// VectorQuantizer_85023172591762
// MI455X (gfx1250) — matrix operand order repaired and checked by reading, not compiled, not run on hardware
//
#include <hip/hip_runtime.h>

// ---------------------------------------------------------------------------
// VQ (cosine/L2-normalized) quantizer for MI455X (gfx1250, wave32, WMMA).
// Distances GEMM runs on v_wmma_f32_16x16x32_f16 (f32 accumulate) with
// register double-buffered B fragments; the whole 16-row x 4096-col distance
// tile lives in 262KB LDS (CDNA5: 320KB/WGP) so softmax + argmin + prob
// writes happen in a single fused pass over HBM.
// ---------------------------------------------------------------------------

typedef _Float16 h4   __attribute__((ext_vector_type(4)));
typedef _Float16 h8   __attribute__((ext_vector_type(8)));
typedef _Float16 v16h __attribute__((ext_vector_type(16)));
typedef float    v8f  __attribute__((ext_vector_type(8)));

// 4-byte-aligned float4 wrapper: tuple output offsets are odd (e.g. +1), so
// global float4 stores are only dword-aligned.
struct __attribute__((packed)) f4p { float4 v; };

#define DIM   256
#define KCB   4096
#define NROWS 32768
#define SD    4100   // LDS row stride (dwords): mod64==4 (bank skew), mod4==0 (b128 ok)

static constexpr size_t OFF_VQ   = 0;
static constexpr size_t OFF_Q    = 1;
static constexpr size_t OFF_P    = OFF_Q + (size_t)NROWS * DIM;   // 8388609
static constexpr size_t OFF_PERP = OFF_P + (size_t)NROWS * KCB;   // 142606337
static constexpr size_t OFF_IDX  = OFF_PERP + 1;                  // 142606338
static constexpr size_t OFF_ACT  = OFF_IDX + NROWS;               // 142639106
static constexpr size_t OFF_USE  = OFF_ACT + 1;                   // 142639107

// ---------------------------------------------------------------------------
__global__ void vq_zero(float* __restrict__ counts, float* __restrict__ dsum) {
  int t = blockIdx.x * blockDim.x + threadIdx.x;
  if (t < KCB) counts[t] = 0.0f;
  if (t == 0)  dsum[0]   = 0.0f;
}

// ---------------------------------------------------------------------------
// One wave32 per row: l2-normalize, emit f32 (optional) + f16 copies.
__global__ __launch_bounds__(256)
void vq_normalize(const float* __restrict__ in, float* __restrict__ outf,
                  _Float16* __restrict__ outh, int nrows) {
  const int row = blockIdx.x * (blockDim.x >> 5) + (threadIdx.x >> 5);
  if (row >= nrows) return;
  const int lane = threadIdx.x & 31;

  const float4* p = (const float4*)(in + (size_t)row * DIM);
  float4 x0 = p[lane];
  float4 x1 = p[lane + 32];
  float s = x0.x*x0.x + x0.y*x0.y + x0.z*x0.z + x0.w*x0.w
          + x1.x*x1.x + x1.y*x1.y + x1.z*x1.z + x1.w*x1.w;
#pragma unroll
  for (int m = 16; m >= 1; m >>= 1) s += __shfl_xor(s, m, 32);

  const float inv = 1.0f / fmaxf(sqrtf(s), 1e-12f);
  x0.x *= inv; x0.y *= inv; x0.z *= inv; x0.w *= inv;
  x1.x *= inv; x1.y *= inv; x1.z *= inv; x1.w *= inv;

  if (outf) {
    float4* q = (float4*)(outf + (size_t)row * DIM);
    q[lane]      = x0;
    q[lane + 32] = x1;
  }
  h4 hh0 = { (_Float16)x0.x, (_Float16)x0.y, (_Float16)x0.z, (_Float16)x0.w };
  h4 hh1 = { (_Float16)x1.x, (_Float16)x1.y, (_Float16)x1.z, (_Float16)x1.w };
  h4* qh = (h4*)(outh + (size_t)row * DIM);
  qh[lane]      = hh0;
  qh[lane + 32] = hh1;
}

// ---------------------------------------------------------------------------
// Fused: WMMA distances (16 rows x 4096 codes) -> LDS tile -> softmax sum,
// argmin, counts/loss atomics, codebook gather, normalized prob writes.
__global__ __launch_bounds__(256)
void vq_fused(const _Float16* __restrict__ flat_h, const _Float16* __restrict__ cb_h,
              const float* __restrict__ cb_f, float* __restrict__ counts,
              float* __restrict__ dsum, float* __restrict__ out) {
  extern __shared__ float dist[];              // 16 * SD floats (262400 B)
  __shared__ float sSum[16][17];
  __shared__ float sMin[16][17];
  __shared__ int   sArg[16][17];
  __shared__ float rowInv[16];
  __shared__ int   rowArg[16];

  const int tid  = threadIdx.x;
  const int lane = tid & 31;
  const int wave = tid >> 5;
  const int hi   = lane >> 4;          // which 16-lane half
  const int l15  = lane & 15;
  const int row0 = blockIdx.x * 16;

  // --- A fragments: 16x256 f16 tile resident in registers (contiguous K order,
  //     matching B: lane half selects K sub-block; elements 0..15 -> K=kb+16*hi..)
  const _Float16* Arow = flat_h + (size_t)(row0 + l15) * DIM;
  v16h a[8];
#pragma unroll
  for (int kk = 0; kk < 8; ++kk) {
    const int kb = kk * 32;
    a[kk] = *(const v16h*)(Arow + kb + 16 * hi);
  }

  // --- GEMM with register double-buffered B fragments. Each wave owns a
  //     16-column slab per pass; passes ping-pong b0/b1 so the 16 b128 loads
  //     of the next pass are in flight behind the 8 WMMAs of the current one.
  const _Float16* Bbase = cb_h + (size_t)(wave * 16 + l15) * DIM + 16 * hi;
  const size_t    Bstep = (size_t)128 * DIM;   // 128 columns per pass

  v16h b0[8], b1[8];
#pragma unroll
  for (int kk = 0; kk < 8; ++kk) b0[kk] = *(const v16h*)(Bbase + kk * 32);

#pragma unroll 1
  for (int pp = 0; pp < 16; ++pp) {
    const int p0 = 2 * pp, p1 = 2 * pp + 1;

    // prefetch pass p1
    const _Float16* Bp1 = Bbase + (size_t)p1 * Bstep;
#pragma unroll
    for (int kk = 0; kk < 8; ++kk) b1[kk] = *(const v16h*)(Bp1 + kk * 32);

    // compute pass p0 on b0
    {
      v8f c = {};
#pragma unroll
      for (int kk = 0; kk < 8; ++kk)
        c = __builtin_amdgcn_wmma_f32_16x16x32_f16(false, a[kk], false, b0[kk],
                                                   (short)0, c, false, false);
      const int n = p0 * 128 + wave * 16 + l15;
#pragma unroll
      for (int r = 0; r < 8; ++r)
        dist[(hi * 8 + r) * SD + n] = 2.0f - 2.0f * c[r];
    }

    // prefetch pass p0+2 (next iteration's b0)
    if (pp < 15) {
      const _Float16* Bp2 = Bbase + (size_t)(p0 + 2) * Bstep;
#pragma unroll
      for (int kk = 0; kk < 8; ++kk) b0[kk] = *(const v16h*)(Bp2 + kk * 32);
    }

    // compute pass p1 on b1
    {
      v8f c = {};
#pragma unroll
      for (int kk = 0; kk < 8; ++kk)
        c = __builtin_amdgcn_wmma_f32_16x16x32_f16(false, a[kk], false, b1[kk],
                                                   (short)0, c, false, false);
      const int n = p1 * 128 + wave * 16 + l15;
#pragma unroll
      for (int r = 0; r < 8; ++r)
        dist[(hi * 8 + r) * SD + n] = 2.0f - 2.0f * c[r];
    }
  }
  __syncthreads();

  // --- per-row expsum + argmin; b128 LDS reads, 16 slices/row, groups
  //     interleaved so per-thread column order stays ascending (first-min ties)
  {
    const int r  = tid & 15;
    const int s0 = tid >> 4;
    const float4* drow = (const float4*)(dist + r * SD);
    float mn = 3.0e38f, se = 0.0f;
    int arg = 0;
    for (int j = 0; j < 64; ++j) {
      const int g   = s0 + (j << 4);          // float4 group 0..1023
      const float4 d4 = drow[g];
      const int cb  = g << 2;
      se += __expf(-d4.x) + __expf(-d4.y) + __expf(-d4.z) + __expf(-d4.w);
      if (d4.x < mn) { mn = d4.x; arg = cb;     }
      if (d4.y < mn) { mn = d4.y; arg = cb + 1; }
      if (d4.z < mn) { mn = d4.z; arg = cb + 2; }
      if (d4.w < mn) { mn = d4.w; arg = cb + 3; }
    }
    sSum[r][s0] = se; sMin[r][s0] = mn; sArg[r][s0] = arg;
  }
  __syncthreads();

  if (tid < 16) {
    const int rr = tid;
    float S = 0.0f, m = 3.0e38f;
    int am = 0;
    for (int s = 0; s < 16; ++s) {
      S += sSum[rr][s];
      const float mv = sMin[rr][s];
      const int   av = sArg[rr][s];
      if (mv < m || (mv == m && av < am)) { m = mv; am = av; }  // jnp first-min ties
    }
    rowInv[rr] = 1.0f / S;
    rowArg[rr] = am;
    out[OFF_IDX + row0 + rr] = (float)am;
    atomicAdd(&counts[am], 1.0f);
    atomicAdd(dsum, m);                               // sum of min distances
  }
  __syncthreads();

  // --- quantized_st rows = normalized codebook rows (forward value)
  for (int i = tid; i < 16 * (DIM / 4); i += 256) {   // 1024 float4 groups
    const int rr = i >> 6;
    const int g  = i & 63;
    const float4 v = ((const float4*)(cb_f + (size_t)rowArg[rr] * DIM))[g];
    ((f4p*)(out + OFF_Q + (size_t)(row0 + rr) * DIM))[g].v = v;
  }

  // --- soft_probs = exp(-d) / sum  (537MB stream: the HBM-bound part)
  for (int i = tid; i < 16 * (KCB / 4); i += 256) {   // 16384 float4 groups
    const int rr = i >> 10;
    const int g  = i & 1023;
    const float4 d4 = *(const float4*)(dist + rr * SD + (g << 2));
    const float inv = rowInv[rr];
    float4 p4;
    p4.x = __expf(-d4.x) * inv;
    p4.y = __expf(-d4.y) * inv;
    p4.z = __expf(-d4.z) * inv;
    p4.w = __expf(-d4.w) * inv;
    ((f4p*)(out + OFF_P + (size_t)(row0 + rr) * KCB))[g].v = p4;
  }
}

// ---------------------------------------------------------------------------
__global__ __launch_bounds__(256)
void vq_finalize(const float* __restrict__ counts, const float* __restrict__ dsum,
                 float* __restrict__ out) {
  __shared__ float sE[256];
  __shared__ float sA[256];
  const int tid = threadIdx.x;
  float ent = 0.0f, act = 0.0f;
  for (int k = tid; k < KCB; k += 256) {
    const float c   = counts[k];
    const float avg = c * (1.0f / (float)NROWS);
    ent += avg * logf(avg + 1e-10f);
    act += (c > 0.0f) ? 1.0f : 0.0f;
  }
  sE[tid] = ent; sA[tid] = act;
  __syncthreads();
  for (int s = 128; s > 0; s >>= 1) {
    if (tid < s) { sE[tid] += sE[tid + s]; sA[tid] += sA[tid + s]; }
    __syncthreads();
  }
  if (tid == 0) {
    out[OFF_PERP] = __expf(-sE[0]);
    out[OFF_ACT]  = sA[0];
    out[OFF_USE]  = sA[0] * (100.0f / (float)KCB);
    out[OFF_VQ]   = 0.25f * dsum[0] / (float)((size_t)NROWS * DIM);
  }
}

// ---------------------------------------------------------------------------
extern "C" void kernel_launch(void* const* d_in, const int* in_sizes, int n_in,
                              void* d_out, int out_size, void* d_ws, size_t ws_size,
                              hipStream_t stream) {
  (void)in_sizes; (void)n_in; (void)out_size; (void)ws_size;
  const float* inputs = (const float*)d_in[0];   // (16,2048,256) f32
  const float* weight = (const float*)d_in[1];   // (4096,256)    f32
  float* out = (float*)d_out;

  char* ws = (char*)d_ws;
  float*    cb_f   = (float*)    (ws + 0);          //  4 MB  normalized codebook f32
  _Float16* cb_h   = (_Float16*) (ws + 4194304);    //  2 MB  normalized codebook f16
  _Float16* flat_h = (_Float16*) (ws + 6291456);    // 16 MB  normalized inputs  f16
  float*    counts = (float*)    (ws + 23068672);   // 16 KB
  float*    dsumv  = (float*)    (ws + 23085056);   //  4 B

  vq_zero<<<16, 256, 0, stream>>>(counts, dsumv);
  vq_normalize<<<KCB / 8,   256, 0, stream>>>(weight, cb_f, cb_h, KCB);
  vq_normalize<<<NROWS / 8, 256, 0, stream>>>(inputs, nullptr, flat_h, NROWS);
  vq_fused<<<NROWS / 16, 256, 16 * SD * sizeof(float), stream>>>(
      flat_h, cb_h, cb_f, counts, dsumv, out);
  vq_finalize<<<1, 256, 0, stream>>>(counts, dsumv, out);
}
